// QHNet_67010079752580
// MI455X (gfx1250) — matrix operand order repaired and checked by reading, not compiled, not run on hardware
//
#include <hip/hip_runtime.h>
#include <hip/hip_bf16.h>
#include <math.h>

#define NN 20000   // multiple of 16: tiles cover exactly, no bounds guards needed
#define EE 200000  // multiple of 16
#define DD 512

static_assert(NN % 16 == 0 && EE % 16 == 0, "tile-exact sizes assumed");

typedef _Float16 h16;
typedef __attribute__((ext_vector_type(16))) _Float16 v16h;
typedef __attribute__((ext_vector_type(8)))  float    v8f;

__device__ __forceinline__ v8f wmma_f16(v16h a, v16h b, v8f c) {
  // D = A(16x32 f16) * B(32x16 f16) + C(16x16 f32)
  return __builtin_amdgcn_wmma_f32_16x16x32_f16(false, a, false, b, (short)0, c, false, false);
}

// A fragment (16x32 slice at column k0) from LDS, row-major, row stride lda (halfs).
// Contiguous k order (matches k_pack): lane m = lane&15; elems 0..15 -> K = k0 + (lane>>4)*16 + j.
__device__ __forceinline__ v16h load_a(const h16* __restrict__ A, int lda, int k0, int lane) {
  const int m  = lane & 15;
  const int kb = k0 + ((lane >> 4) << 4);
  v16h a;
#pragma unroll
  for (int j = 0; j < 8; ++j) {
    a[j]     = A[m * lda + kb + j];
    a[j + 8] = A[m * lda + kb + 8 + j];
  }
  return a;
}

// B fragment from packed weights: [nt][kt][lane][16 halfs], 32B/lane contiguous.
__device__ __forceinline__ v16h load_bp(const h16* __restrict__ P, int nkt, int k0, int nb, int lane) {
  const int kt = k0 >> 5, nt = nb >> 4;
  return *(const v16h*)(P + (((size_t)(nt * nkt + kt) * 32 + lane) << 4));
}

__device__ __forceinline__ float silu_f(float v) { return v / (1.f + __expf(-v)); }
__device__ __forceinline__ float ssp_f(float v) {
  return fmaxf(v, 0.f) + log1pf(__expf(-fabsf(v))) - 0.69314718055994531f;
}
__device__ __forceinline__ void atomic_add_f(float* p, float v) {
  unsafeAtomicAdd(p, v);  // native global_atomic_add_f32 (accumulator lives in L2)
}

// ---------------------------------------------------------------------------
// K0: pack f32 weight (K x N row-major) into f16 WMMA-B fragment layout:
//     P[((nt*nkt + kt)*32 + lane)*16 + j] = W[(kt*32 + (lane>>4)*16 + j)*N + nt*16 + (lane&15)]
// ---------------------------------------------------------------------------
__global__ __launch_bounds__(256) void k_pack(const float* __restrict__ W, int K, int N,
                                              h16* __restrict__ P) {
  int idx = blockIdx.x * 256 + threadIdx.x;
  if (idx >= K * N) return;
  int j    = idx & 15;
  int lane = (idx >> 4) & 31;
  int t    = idx >> 9;
  int nkt  = K >> 5;
  int kt   = t % nkt;
  int nt   = t / nkt;
  int k = (kt << 5) + ((lane >> 4) << 4) + j;
  int n = (nt << 4) + (lane & 15);
  P[idx] = (h16)W[k * N + n];
}

// ---------------------------------------------------------------------------
// K1: pre = irrep_linear(x, W_pre0,b_pre0,W_pre1) -> p0 (Nx128), p1 (Nx384 interleaved)
// grid.x = node tiles, grid.y = 0:scalar, 1..3: vector component
// ---------------------------------------------------------------------------
__global__ __launch_bounds__(256) void k_pre(const float* __restrict__ x,
    const h16* __restrict__ P0, const float* __restrict__ b0,
    const h16* __restrict__ P1, float* __restrict__ p0, float* __restrict__ p1) {
  __shared__ h16 As[16 * 128];
  const int base = blockIdx.x * 16;
  const int comp = blockIdx.y;
  const int tid  = threadIdx.x;
  for (int idx = tid; idx < 16 * 128; idx += 256) {
    int m = idx >> 7, k = idx & 127;
    int row = base + m;
    float v = (comp == 0) ? x[row * DD + k] : x[row * DD + 128 + 3 * k + (comp - 1)];
    As[idx] = (h16)v;
  }
  __syncthreads();
  const int wave = tid >> 5, lane = tid & 31;
  const int nb = wave * 16;
  const h16* P = (comp == 0) ? P0 : P1;
  v8f acc = {};
  for (int k0 = 0; k0 < 128; k0 += 32)
    acc = wmma_f16(load_a(As, 128, k0, lane), load_bp(P, 4, k0, nb, lane), acc);
  const int n = nb + (lane & 15);
  if (comp == 0) {
    const float bias = b0[n];
#pragma unroll
    for (int r = 0; r < 8; ++r) {
      int row = base + ((lane >> 4) << 3) + r;
      p0[row * 128 + n] = acc[r] + bias;
    }
  } else {
#pragma unroll
    for (int r = 0; r < 8; ++r) {
      int row = base + ((lane >> 4) << 3) + r;
      p1[row * 384 + 3 * n + (comp - 1)] = acc[r];
    }
  }
}

// ---------------------------------------------------------------------------
// K2: gate + node-update linear, fully fused per 16-node tile -> self_x (N x 512)
// ---------------------------------------------------------------------------
__global__ __launch_bounds__(256) void k_gate_nd(const float* __restrict__ x,
    const h16* __restrict__ Pg1, const float* __restrict__ bg1,
    const h16* __restrict__ Pg2, const float* __restrict__ bg2,
    const h16* __restrict__ Pnd0, const float* __restrict__ bnd0,
    const h16* __restrict__ Pnd1, float* __restrict__ selfx) {
  __shared__ float x1f[16 * 384];     // 24 KB  vector part of x, f32
  __shared__ float gw[16 * 256];      // 16 KB  gate output g, f32
  __shared__ h16   bufA[16 * 256];    //  8 KB  ag  -> later an[0], an[1]
  __shared__ h16   bufB[16 * 256];    //  8 KB  h   -> later an[2], an[3]
  const int base = blockIdx.x * 16;
  const int tid  = threadIdx.x;
  const int wave = tid >> 5, lane = tid & 31;

  for (int idx = tid; idx < 16 * 512; idx += 256) {
    int m = idx >> 9, k = idx & 511;
    float v = x[(base + m) * DD + k];
    if (k < 128) bufA[m * 256 + k] = (h16)v;          // ag[:, :128] = x0
    else         x1f[m * 384 + (k - 128)] = v;
  }
  __syncthreads();
  for (int idx = tid; idx < 16 * 128; idx += 256) {   // ag[:, 128:] = |x1| norms
    int m = idx >> 7, u = idx & 127;
    float a = x1f[m * 384 + 3 * u], b = x1f[m * 384 + 3 * u + 1], c = x1f[m * 384 + 3 * u + 2];
    bufA[m * 256 + 128 + u] = (h16)sqrtf((a * a + b * b + c * c) * (1.f / 3.f));
  }
  __syncthreads();
  for (int t = wave; t < 16; t += 8) {                // h = silu(ag @ Wg1 + bg1)
    int nb = t * 16;
    v8f acc = {};
    for (int k0 = 0; k0 < 256; k0 += 32)
      acc = wmma_f16(load_a(bufA, 256, k0, lane), load_bp(Pg1, 8, k0, nb, lane), acc);
    int n = nb + (lane & 15);
    float bias = bg1[n];
#pragma unroll
    for (int r = 0; r < 8; ++r)
      bufB[(((lane >> 4) << 3) + r) * 256 + n] = (h16)silu_f(acc[r] + bias);
  }
  __syncthreads();
  for (int t = wave; t < 16; t += 8) {                // g = h @ Wg2 + bg2
    int nb = t * 16;
    v8f acc = {};
    for (int k0 = 0; k0 < 256; k0 += 32)
      acc = wmma_f16(load_a(bufB, 256, k0, lane), load_bp(Pg2, 8, k0, nb, lane), acc);
    int n = nb + (lane & 15);
    float bias = bg2[n];
#pragma unroll
    for (int r = 0; r < 8; ++r)
      gw[(((lane >> 4) << 3) + r) * 256 + n] = acc[r] + bias;
  }
  __syncthreads();
  for (int idx = tid; idx < 16 * 128; idx += 256) {   // xg: overlay an[0..3] on bufA/bufB
    int m = idx >> 7, u = idx & 127;
    bufA[m * 128 + u] = (h16)gw[m * 256 + u];                        // an0 = g[:, :128]
    float gv = gw[m * 256 + 128 + u];
    bufA[2048 + m * 128 + u] = (h16)(x1f[m * 384 + 3 * u + 0] * gv); // an1
    bufB[m * 128 + u]        = (h16)(x1f[m * 384 + 3 * u + 1] * gv); // an2
    bufB[2048 + m * 128 + u] = (h16)(x1f[m * 384 + 3 * u + 2] * gv); // an3
  }
  __syncthreads();
  for (int t = wave; t < 32; t += 8) {                // self_x = irrep_linear(xg, Wnd*)
    int comp = t >> 3, nb = (t & 7) * 16;
    const h16* A = (comp == 0) ? bufA : (comp == 1) ? bufA + 2048
                 : (comp == 2) ? bufB : bufB + 2048;
    const h16* P = (comp == 0) ? Pnd0 : Pnd1;
    v8f acc = {};
    for (int k0 = 0; k0 < 128; k0 += 32)
      acc = wmma_f16(load_a(A, 128, k0, lane), load_bp(P, 4, k0, nb, lane), acc);
    int n = nb + (lane & 15);
    if (comp == 0) {
      float bias = bnd0[n];
#pragma unroll
      for (int r = 0; r < 8; ++r) {
        int row = base + ((lane >> 4) << 3) + r;
        selfx[row * DD + n] = acc[r] + bias;
      }
    } else {
#pragma unroll
      for (int r = 0; r < 8; ++r) {
        int row = base + ((lane >> 4) << 3) + r;
        selfx[row * DD + 128 + 3 * n + (comp - 1)] = acc[r];
      }
    }
  }
}

// ---------------------------------------------------------------------------
// K3: edge phase: per 16-edge tile build s0, run both 2-stage MLPs on WMMA,
//     form edge features, scatter with f32 atomics into accumulator (in L2).
// ---------------------------------------------------------------------------
__global__ __launch_bounds__(256) void k_edge(
    const float* __restrict__ x, const float* __restrict__ esh,
    const float* __restrict__ eattr, const int* __restrict__ eidx,
    const float* __restrict__ p0, const float* __restrict__ p1,
    const h16* __restrict__ Pf1, const h16* __restrict__ Pf2,
    const h16* __restrict__ Ps1, const h16* __restrict__ Ps2,
    float* __restrict__ accum) {
  __shared__ h16   Ae[16 * 32];       //  1 KB
  __shared__ h16   As[16 * 384];      // 12 KB  s0 = [pd0, pd0, ip1]
  __shared__ h16   t1h[16 * 32];      //  1 KB
  __shared__ h16   t2h[16 * 32];      //  1 KB
  __shared__ float wb[16 * 512];      // 32 KB  w = wA * wB
  __shared__ float shs[16 * 4];
  __shared__ int   dsts[16], srcs[16];
  const int base = blockIdx.x * 16;
  const int tid  = threadIdx.x;
  const int wave = tid >> 5, lane = tid & 31;

  // pull the small packed weight tables through GL2 ahead of the GEMM stages
  __builtin_prefetch(Pf2 + (size_t)(tid << 4), 0, 3);
  __builtin_prefetch(Ps2 + (size_t)(tid << 4), 0, 3);

  if (tid < 16) {
    int e = base + tid;
    dsts[tid] = eidx[e];
    srcs[tid] = eidx[EE + e];
  }
  if (tid < 64) {
    int e = base + (tid >> 2);
    shs[tid] = esh[e * 4 + (tid & 3)];
  }
  for (int idx = tid; idx < 16 * 32; idx += 256) {
    int m = idx >> 5, k = idx & 31;
    Ae[idx] = (h16)eattr[(base + m) * 32 + k];
  }
  __syncthreads();
  for (int idx = tid; idx < 16 * 128; idx += 256) {   // gather p rows, build s0
    int m = idx >> 7, u = idx & 127;
    int d = dsts[m], s = srcs[m];
    float pv = p0[d * 128 + u];
    float ip = (p1[d * 384 + 3 * u + 0] * p1[s * 384 + 3 * u + 0] +
                p1[d * 384 + 3 * u + 1] * p1[s * 384 + 3 * u + 1] +
                p1[d * 384 + 3 * u + 2] * p1[s * 384 + 3 * u + 2]) * (1.f / 3.f);
    h16 ph = (h16)pv;
    As[m * 384 + u]       = ph;
    As[m * 384 + 128 + u] = ph;
    As[m * 384 + 256 + u] = (h16)ip;
  }
  __syncthreads();
  if (wave < 2) {                                     // t1 = ssp(eattr @ Wf1), K=32
    int nb = wave * 16;
    v8f acc = {};
    acc = wmma_f16(load_a(Ae, 32, 0, lane), load_bp(Pf1, 1, 0, nb, lane), acc);
    int n = nb + (lane & 15);
#pragma unroll
    for (int r = 0; r < 8; ++r)
      t1h[(((lane >> 4) << 3) + r) * 32 + n] = (h16)ssp_f(acc[r]);
  } else if (wave < 4) {                              // t2 = ssp(s0 @ Ws1), K=384
    int nb = (wave - 2) * 16;
    v8f acc = {};
    for (int k0 = 0; k0 < 384; k0 += 32)
      acc = wmma_f16(load_a(As, 384, k0, lane), load_bp(Ps1, 12, k0, nb, lane), acc);
    int n = nb + (lane & 15);
#pragma unroll
    for (int r = 0; r < 8; ++r)
      t2h[(((lane >> 4) << 3) + r) * 32 + n] = (h16)ssp_f(acc[r]);
  } else {
    // waves 4..7: prefetch the gathered x[edge_src] rows (2 KB each) that the
    // scatter phase needs, overlapping random-gather latency with the GEMMs.
    int t2i = tid - 128;                 // 0..127
    int m2  = t2i >> 3;                  // edge slot 0..15
    int seg = t2i & 7;                   // 0..7, 256 B each -> full 2 KB row
    const char* xp = (const char*)(x + (size_t)srcs[m2] * DD);
    __builtin_prefetch(xp + seg * 256, 0, 0);
  }
  __syncthreads();
  for (int t = wave; t < 32; t += 8) {                // w = (t1 @ Wf2) * (t2 @ Ws2)
    int nb = t * 16;
    v16h a1 = load_a(t1h, 32, 0, lane);
    v16h a2 = load_a(t2h, 32, 0, lane);
    v8f accA = {}, accB = {};
    accA = wmma_f16(a1, load_bp(Pf2, 1, 0, nb, lane), accA);
    accB = wmma_f16(a2, load_bp(Ps2, 1, 0, nb, lane), accB);
    int n = nb + (lane & 15);
#pragma unroll
    for (int r = 0; r < 8; ++r)
      wb[(((lane >> 4) << 3) + r) * 512 + n] = accA[r] * accB[r];
  }
  __syncthreads();
  const float inv_sqrt3 = 0.57735026918962576f;
  const int m  = tid >> 4;
  const int u0 = (tid & 15) * 8;
  {
    int d = dsts[m], s = srcs[m];
    float sh0 = shs[m * 4];
    float s1x = shs[m * 4 + 1], s1y = shs[m * 4 + 2], s1z = shs[m * 4 + 3];
    const float* xs = x + (size_t)s * DD;
    float* ao = accum + (size_t)d * DD;
#pragma unroll
    for (int j = 0; j < 8; ++j) {
      int u = u0 + j;
      float w1 = wb[m * 512 + u];
      float w2 = wb[m * 512 + 128 + u];
      float w3 = wb[m * 512 + 256 + u];
      float w4 = wb[m * 512 + 384 + u];
      float x0v = xs[u];
      float xa = xs[128 + 3 * u], xbv = xs[128 + 3 * u + 1], xc = xs[128 + 3 * u + 2];
      float dotv = xa * s1x + xbv * s1y + xc * s1z;
      atomic_add_f(ao + u, w1 * x0v * sh0 + w4 * dotv * inv_sqrt3);
      atomic_add_f(ao + 128 + 3 * u + 0, w2 * x0v * s1x + w3 * xa  * sh0);
      atomic_add_f(ao + 128 + 3 * u + 1, w2 * x0v * s1y + w3 * xbv * sh0);
      atomic_add_f(ao + 128 + 3 * u + 2, w2 * x0v * s1z + w3 * xc  * sh0);
    }
  }
}

// ---------------------------------------------------------------------------
// K4: out = irrep_linear(acc + self_x, Wo*) + old_x
// ---------------------------------------------------------------------------
__global__ __launch_bounds__(256) void k_out(const float* __restrict__ accum,
    const float* __restrict__ selfx, const float* __restrict__ xold,
    const h16* __restrict__ P0, const float* __restrict__ b0,
    const h16* __restrict__ P1, float* __restrict__ out) {
  __shared__ h16 As[16 * 128];
  const int base = blockIdx.x * 16;
  const int comp = blockIdx.y;
  const int tid  = threadIdx.x;
  for (int idx = tid; idx < 16 * 128; idx += 256) {
    int m = idx >> 7, k = idx & 127;
    int row = base + m;
    int col = (comp == 0) ? k : 128 + 3 * k + (comp - 1);
    As[idx] = (h16)(accum[row * DD + col] + selfx[row * DD + col]);
  }
  __syncthreads();
  const int wave = tid >> 5, lane = tid & 31;
  const int nb = wave * 16;
  const h16* P = (comp == 0) ? P0 : P1;
  v8f acc = {};
  for (int k0 = 0; k0 < 128; k0 += 32)
    acc = wmma_f16(load_a(As, 128, k0, lane), load_bp(P, 4, k0, nb, lane), acc);
  const int n = nb + (lane & 15);
  if (comp == 0) {
    const float bias = b0[n];
#pragma unroll
    for (int r = 0; r < 8; ++r) {
      int row = base + ((lane >> 4) << 3) + r;
      out[row * DD + n] = acc[r] + bias + xold[row * DD + n];
    }
  } else {
#pragma unroll
    for (int r = 0; r < 8; ++r) {
      int row = base + ((lane >> 4) << 3) + r;
      int col = 128 + 3 * n + (comp - 1);
      out[row * DD + col] = acc[r] + xold[row * DD + col];
    }
  }
}

extern "C" void kernel_launch(void* const* d_in, const int* in_sizes, int n_in,
                              void* d_out, int out_size, void* d_ws, size_t ws_size,
                              hipStream_t stream) {
  const float* x     = (const float*)d_in[0];
  const float* esh   = (const float*)d_in[1];
  const float* eattr = (const float*)d_in[2];
  const int*   eidx  = (const int*)d_in[3];
  const float* Wpre0 = (const float*)d_in[4];
  const float* bpre0 = (const float*)d_in[5];
  const float* Wpre1 = (const float*)d_in[6];
  const float* Wnd0  = (const float*)d_in[7];
  const float* bnd0  = (const float*)d_in[8];
  const float* Wnd1  = (const float*)d_in[9];
  const float* Wg1   = (const float*)d_in[10];
  const float* bg1   = (const float*)d_in[11];
  const float* Wg2   = (const float*)d_in[12];
  const float* bg2   = (const float*)d_in[13];
  const float* Wf1   = (const float*)d_in[14];
  const float* Wf2   = (const float*)d_in[15];
  const float* Ws1   = (const float*)d_in[16];
  const float* Ws2   = (const float*)d_in[17];
  const float* Wo0   = (const float*)d_in[18];
  const float* bo0   = (const float*)d_in[19];
  const float* Wo1   = (const float*)d_in[20];

  float* ws    = (float*)d_ws;
  float* p0    = ws;                          // N*128 f32
  float* p1    = p0 + (size_t)NN * 128;       // N*384 f32
  float* selfx = p1 + (size_t)NN * 384;       // N*512 f32
  float* accum = selfx + (size_t)NN * 512;    // N*512 f32

  h16* pk     = (h16*)(accum + (size_t)NN * 512);
  h16* Ppre0  = pk;              pk += 128 * 128;
  h16* Ppre1  = pk;              pk += 128 * 128;
  h16* Pnd0   = pk;              pk += 128 * 128;
  h16* Pnd1   = pk;              pk += 128 * 128;
  h16* Pg1    = pk;              pk += 256 * 256;
  h16* Pg2    = pk;              pk += 256 * 256;
  h16* Pf1    = pk;              pk += 32 * 32;
  h16* Pf2    = pk;              pk += 32 * 512;
  h16* Ps1    = pk;              pk += 384 * 32;
  h16* Ps2    = pk;              pk += 32 * 512;
  h16* Po0    = pk;              pk += 128 * 128;
  h16* Po1    = pk;              pk += 128 * 128;

  hipMemsetAsync(accum, 0, (size_t)NN * 512 * sizeof(float), stream);

  dim3 blk(256);
  auto pack = [&](const float* W, int K, int N, h16* P) {
    k_pack<<<dim3((K * N + 255) / 256), blk, 0, stream>>>(W, K, N, P);
  };
  pack(Wpre0, 128, 128, Ppre0);  pack(Wpre1, 128, 128, Ppre1);
  pack(Wnd0, 128, 128, Pnd0);    pack(Wnd1, 128, 128, Pnd1);
  pack(Wg1, 256, 256, Pg1);      pack(Wg2, 256, 256, Pg2);
  pack(Wf1, 32, 32, Pf1);        pack(Wf2, 32, 512, Pf2);
  pack(Ws1, 384, 32, Ps1);       pack(Ws2, 32, 512, Ps2);
  pack(Wo0, 128, 128, Po0);      pack(Wo1, 128, 128, Po1);

  k_pre    <<<dim3(NN / 16, 4), blk, 0, stream>>>(x, Ppre0, bpre0, Ppre1, p0, p1);
  k_gate_nd<<<dim3(NN / 16),    blk, 0, stream>>>(x, Pg1, bg1, Pg2, bg2,
                                                  Pnd0, bnd0, Pnd1, selfx);
  k_edge   <<<dim3(EE / 16),    blk, 0, stream>>>(x, esh, eattr, eidx, p0, p1,
                                                  Pf1, Pf2, Ps1, Ps2, accum);
  k_out    <<<dim3(NN / 16, 4), blk, 0, stream>>>(accum, selfx, x, Po0, bo0, Po1,
                                                  (float*)d_out);
}
